// DAMPS_65085934403885
// MI455X (gfx1250) — matrix operand order repaired and checked by reading, not compiled, not run on hardware
//
#include <hip/hip_runtime.h>
#include <hip/hip_bf16.h>
#include <math.h>

typedef __bf16 bf16_t;
typedef __attribute__((ext_vector_type(4)))  bf16_t v4bf;
typedef __attribute__((ext_vector_type(8)))  bf16_t v8bf;
typedef __attribute__((ext_vector_type(16))) bf16_t v16bf;
typedef __attribute__((ext_vector_type(8)))  float  v8f;
typedef __attribute__((ext_vector_type(4)))  unsigned int u32x4;
typedef __attribute__((ext_vector_type(8)))  int i32x8;
typedef __attribute__((ext_vector_type(4)))  int i32x4;

#define N_ITEMS 30000
#define DIM     128
#define NFREQ   65
#define RAW_IMG 1024
#define RAW_TXT 768
#define TILE_M  64
#define P_STRIDE 130     // padded row stride for projection buffers (floats)
#define AB_STRIDE 40     // padded row stride for bf16 staging tiles (elems); 80B rows
#define ASB_ELEMS (64  * AB_STRIDE)   // 2560 elems = 5120 B per A buffer
#define BSB_ELEMS (128 * AB_STRIDE)   // 5120 elems = 10240 B per B buffer

// ---- dynamic LDS layout (bytes) ----
#define PIMG_OFF   0                       // 64*130*4 = 33280
#define PTXT_OFF   33280
#define YIR_OFF    66560                   // 64*65*4 = 16640
#define YII_OFF    83200
#define YTR_OFF    99840
#define YTI_OFF    116480                  // ends 133120
#define AS_OFF     66560                   // overlay on Y (disjoint in time): 2*5120
#define BS_OFF     76800                   // 2*10240, ends 97280 (< YTR_OFF)
#define CT_OFF     133120                  // 128*4
#define ROTC_OFF   133632
#define ROTS_OFF   133904
#define AVI_OFF    134176
#define AVT_OFF    134448
#define SMEM_BYTES 134720

// ---------------------------------------------------------------------------
// Prep: transpose W [K,128] f32 -> WT [128,K] bf16 (tiny; lives in L2 after)
// ---------------------------------------------------------------------------
__global__ void prep_wt(const float* __restrict__ Wi, const float* __restrict__ Wt,
                        bf16_t* __restrict__ WTi, bf16_t* __restrict__ WTt) {
    int e = blockIdx.x * 256 + threadIdx.x;
    const int NI = RAW_IMG * DIM;
    const int NT = RAW_TXT * DIM;
    if (e < NI) {
        int k = e >> 7, n = e & 127;
        WTi[(size_t)n * RAW_IMG + k] = (bf16_t)Wi[e];
    } else if (e < NI + NT) {
        int e2 = e - NI;
        int k = e2 >> 7, n = e2 & 127;
        WTt[(size_t)n * RAW_TXT + k] = (bf16_t)Wt[e2];
    }
}

// ---------------------------------------------------------------------------
// TDM: DMA one B tile (128 rows x 32 K of bf16, row stride = raw elems) into
// LDS, with hardware padding of 4 DWORDs every 16 DWORDs to produce the
// conflict-free 80B LDS row stride. Wave-uniform; issued by one wave.
// D# group0: count=1, lds_addr, global_addr[56:0], type=2.
// D# group1: data_size=2B(code1), pad_enable, pad_interval=16DW(code3),
//            pad_amount=4DW(code3), tensor_dim0=raw, tensor_dim1=128,
//            tile_dim0=32, tile_dim1=128, tensor_dim0_stride=raw.
// Groups 2/3 (+extra group in the 6-arg builtin form) are unused for a 2D
// tile: all-zero fields mean tile_dim3/4 = 0 = "unused".
// ---------------------------------------------------------------------------
__device__ __forceinline__ void tdm_load_b_tile(const bf16_t* gsrc,
                                                unsigned lds_byte, int raw) {
    unsigned long long ga = (unsigned long long)(uintptr_t)gsrc;
    u32x4 g0;
    g0[0] = 1u;                                   // count=1 (valid user D#)
    g0[1] = lds_byte;                             // lds_addr
    g0[2] = (unsigned)ga;                         // global_addr[31:0]
    g0[3] = (unsigned)((ga >> 32) & 0x01FFFFFFull) | 0x80000000u; // [56:32]|type=2
    i32x8 g1;
    g1[0] = (1 << 16) | (1 << 20) | (3 << 22) | (3 << 25);
    g1[1] = (raw & 0xFFFF) << 16;                 // tensor_dim0 lo16 -> bits[63:48]
    g1[2] = (int)(((unsigned)raw >> 16) | (128u << 16)); // dim0 hi | tensor_dim1 lo
    g1[3] = (int)(32u << 16);                     // tile_dim0=32 -> bits[127:112]
    g1[4] = 128;                                  // tile_dim1=128
    g1[5] = raw;                                  // tensor_dim0_stride lo32
    g1[6] = 0;
    g1[7] = 0;
    i32x4 z4 = {0, 0, 0, 0};
    i32x8 z8 = {0, 0, 0, 0, 0, 0, 0, 0};
    __builtin_amdgcn_tensor_load_to_lds(g0, g1, z4, z4, z8, 0);
}

// ---------------------------------------------------------------------------
// A tile staging: 64 rows x 32 K f32 -> bf16, vectorized (float4 -> v4bf).
// ---------------------------------------------------------------------------
__device__ __forceinline__ void load_a_tile(float4 fa[2], const float* __restrict__ table,
                                            int raw, int row0, int tid, int kc) {
    #pragma unroll
    for (int e = 0; e < 2; ++e) {
        int chunk = e * 256 + tid;
        int r = chunk >> 3, c4 = chunk & 7;
        int grow = row0 + r;
        if (grow >= N_ITEMS) grow = N_ITEMS - 1;   // clamp tail (stores guarded later)
        const float* src = table + (size_t)grow * raw + kc + c4 * 4;
        fa[e] = *(const float4*)src;
        if (kc + 64 < raw) __builtin_prefetch(src + 64, 0, 3);  // global_prefetch_b8
    }
}

__device__ __forceinline__ void store_a_tile(bf16_t* As, const float4 fa[2], int tid) {
    #pragma unroll
    for (int e = 0; e < 2; ++e) {
        int chunk = e * 256 + tid;
        int r = chunk >> 3, c4 = chunk & 7;
        v4bf cv;
        cv[0] = (bf16_t)fa[e].x; cv[1] = (bf16_t)fa[e].y;
        cv[2] = (bf16_t)fa[e].z; cv[3] = (bf16_t)fa[e].w;
        *(v4bf*)(As + r * AB_STRIDE + c4 * 4) = cv;   // one ds_store_b64
    }
}

// ---------------------------------------------------------------------------
// Consume one staged tile: per-wave A fragment + 4 B fragments -> 4 WMMAs.
// ---------------------------------------------------------------------------
__device__ __forceinline__ void consume_tile(const bf16_t* As, const bf16_t* Bs,
                                             v8f acc[4], int Moff, int Nbase,
                                             int half, int l15) {
    // A fragment: lane<16 -> M=l15, K in {0..7}u{16..23}; lane>=16 -> {8..15}u{24..31}
    const bf16_t* ap = As + (Moff + l15) * AB_STRIDE + half * 8;
    v8bf a0 = *(const v8bf*)(ap);
    v8bf a1 = *(const v8bf*)(ap + 16);
    v16bf av = __builtin_shufflevector(a0, a1,
                   0,1,2,3,4,5,6,7,8,9,10,11,12,13,14,15);
    #pragma unroll
    for (int t = 0; t < 4; ++t) {
        const bf16_t* bp = Bs + (Nbase + t * 16 + l15) * AB_STRIDE + half * 8;
        v8bf b0 = *(const v8bf*)(bp);
        v8bf b1 = *(const v8bf*)(bp + 16);
        v16bf bv = __builtin_shufflevector(b0, b1,
                       0,1,2,3,4,5,6,7,8,9,10,11,12,13,14,15);
        acc[t] = __builtin_amdgcn_wmma_f32_16x16x32_bf16(
                     false, av, false, bv, (short)0, acc[t], false, false);
    }
}

// ---------------------------------------------------------------------------
// Double-buffered bf16 WMMA GEMM: P[64x128] = table[64 x raw] @ WT^T + bias.
// B tiles via TDM (TENSORcnt), A tiles via vectorized loads; stage k+1 while
// consuming k.
// ---------------------------------------------------------------------------
__device__ __forceinline__ void gemm_project(
    const float*  __restrict__ table, int raw,
    const bf16_t* __restrict__ WT,
    const float*  __restrict__ bias,
    float* P, bf16_t* As, bf16_t* Bs,
    int row0, int tid)
{
    const int lane  = tid & 31;
    const int wv    = tid >> 5;
    const int Moff  = (wv & 3) * 16;
    const int Nbase = (wv >> 2) * 64;
    const int half  = lane >> 4;
    const int l15   = lane & 15;
    const unsigned bsLo = (unsigned)(uintptr_t)Bs;  // LDS byte offset (low 32 bits)

    v8f acc[4] = {};
    float4 fa[2];
    const int nk = raw >> 5;

    // ---- prologue: stage buffer 0 ----
    load_a_tile(fa, table, raw, row0, tid, 0);
    store_a_tile(As, fa, tid);
    if (wv == 0) {
        tdm_load_b_tile(WT, bsLo, raw);
        __builtin_amdgcn_s_wait_tensorcnt(0);
    }
    __syncthreads();

    for (int kci = 0; kci < nk; ++kci) {
        const int  cur  = kci & 1;
        const int  nxt  = cur ^ 1;
        const bool more = (kci + 1) < nk;
        if (more) {
            load_a_tile(fa, table, raw, row0, tid, (kci + 1) << 5);
            if (wv == 0)
                tdm_load_b_tile(WT + (size_t)((kci + 1) << 5),
                                bsLo + (unsigned)nxt * (BSB_ELEMS * 2), raw);
        }
        consume_tile(As + cur * ASB_ELEMS, Bs + cur * BSB_ELEMS,
                     acc, Moff, Nbase, half, l15);
        if (more) store_a_tile(As + nxt * ASB_ELEMS, fa, tid);
        if (wv == 0) __builtin_amdgcn_s_wait_tensorcnt(0);
        __syncthreads();
    }

    // ---- epilogue: C + bias -> LDS projection buffer (conflict-free stride)
    #pragma unroll
    for (int t = 0; t < 4; ++t) {
        int col = Nbase + t * 16 + l15;
        float bv = bias[col];
        #pragma unroll
        for (int r = 0; r < 8; ++r) {
            P[(Moff + r + half * 8) * P_STRIDE + col] = acc[t][r] + bv;
        }
    }
}

// ---------------------------------------------------------------------------
// Fused kernel: GEMM(img) -> GEMM(txt) -> rDFT -> rotate/MSC/fuse -> irDFT
// ---------------------------------------------------------------------------
__global__ __launch_bounds__(256) void damps_fused(
    const float* __restrict__ img_table, const float* __restrict__ txt_table,
    const bf16_t* __restrict__ WTi, const bf16_t* __restrict__ WTt,
    const float* __restrict__ b_img, const float* __restrict__ b_txt,
    const float* __restrict__ avrf_img, const float* __restrict__ avrf_txt,
    const float* __restrict__ avg_R, const float* __restrict__ psi,
    const float* __restrict__ lw, float* __restrict__ out)
{
    extern __shared__ __align__(16) char smem[];
    float*  Pimg = (float*)(smem + PIMG_OFF);
    float*  Ptxt = (float*)(smem + PTXT_OFF);
    float*  Yir  = (float*)(smem + YIR_OFF);
    float*  Yii  = (float*)(smem + YII_OFF);
    float*  Ytr  = (float*)(smem + YTR_OFF);
    float*  Yti  = (float*)(smem + YTI_OFF);
    bf16_t* As   = (bf16_t*)(smem + AS_OFF);
    bf16_t* Bs   = (bf16_t*)(smem + BS_OFF);
    float*  ct   = (float*)(smem + CT_OFF);
    float*  rotc = (float*)(smem + ROTC_OFF);
    float*  rots = (float*)(smem + ROTS_OFF);
    float*  avi  = (float*)(smem + AVI_OFF);
    float*  avt  = (float*)(smem + AVT_OFF);

    const int tid  = threadIdx.x;
    const int row0 = blockIdx.x * TILE_M;
    const float inv = 0.08838834764831845f;     // 1/sqrt(128) (ortho norm)

    // shared constants: cos table (2*pi*k/128) and per-frequency params.
    if (tid < 128) ct[tid] = cosf((float)tid * 0.049087385212340517f);
    if (tid < NFREQ) {
        float phi = avg_R[tid] * 0.5f + psi[tid];
        rotc[tid] = cosf(phi);
        rots[tid] = sinf(phi);
        avi[tid]  = avrf_img[tid];
        avt[tid]  = avrf_txt[tid];
    }
    // softmax(lambda) in registers (uniform scalar loads)
    float l0 = lw[0], l1 = lw[1];
    float mx = fmaxf(l0, l1);
    float e0 = expf(l0 - mx), e1 = expf(l1 - mx);
    float lam0 = e0 / (e0 + e1), lam1 = 1.0f - lam0;

    // ---- phase 1: both projections via TDM + bf16 WMMA ----
    gemm_project(img_table, RAW_IMG, WTi, b_img, Pimg, As, Bs, row0, tid);
    gemm_project(txt_table, RAW_TXT, WTt, b_txt, Ptxt, As, Bs, row0, tid);
    __syncthreads();

    // ---- phase 2a: forward rDFT + rotation + MSC fusion -> Y spectra ----
    for (int task = tid; task < TILE_M * NFREQ; task += 256) {
        int row = task / NFREQ;
        int f   = task - row * NFREQ;
        const float* pi = Pimg + row * P_STRIDE;
        const float* pt = Ptxt + row * P_STRIDE;
        float xri = 0.f, xii = 0.f, xrt = 0.f, xit = 0.f;
        int idx = 0;                         // (f*d) mod 128, incremental
        #pragma unroll 4
        for (int d = 0; d < DIM; ++d) {
            float cv = ct[idx];
            float sv = ct[(idx + 96) & 127]; // sin(theta) = cos(theta - pi/2)
            float a  = pi[d], b = pt[d];
            xri += a * cv;  xii -= a * sv;   // e^{-i theta}
            xrt += b * cv;  xit -= b * sv;
            idx = (idx + f) & 127;
        }
        xri *= inv; xii *= inv; xrt *= inv; xit *= inv;

        float rc = rotc[f], rs = rots[f];
        // APC_img = X * e^{-i phi};  APC_txt = X * e^{+i phi}
        float air = xri * rc + xii * rs, aii = xii * rc - xri * rs;
        float atr = xrt * rc - xit * rs, ati = xit * rc + xrt * rs;

        float p   = (air * air + aii * aii) * (atr * atr + ati * ati);
        float msc = p / (p + 1e-8f);
        float si  = lam0 * avi[f] + lam1 * msc;
        float st  = lam0 * avt[f] + lam1 * msc;

        int o = row * NFREQ + f;
        Yir[o] = air * si;  Yii[o] = aii * si;
        Ytr[o] = atr * st;  Yti[o] = ati * st;
    }
    __syncthreads();

    // ---- phase 2b: inverse real DFT (numpy irfft semantics) + store ----
    float* outI = out;
    float* outT = out + (size_t)N_ITEMS * DIM;
    for (int task = tid; task < 2 * TILE_M * DIM; task += 256) {
        int m   = task >> 13;                // modality (64*128 = 8192 per)
        int rem = task & 8191;
        int row = rem >> 7;
        int d   = rem & 127;
        const float* Yr = (m ? Ytr : Yir) + row * NFREQ;
        const float* Yi = (m ? Yti : Yii) + row * NFREQ;
        // DC + Nyquist: imaginary parts discarded by irfft
        float acc = Yr[0] + ((d & 1) ? -Yr[64] : Yr[64]);
        int idx = d;                          // (f*d) mod 128 for f=1..
        #pragma unroll 4
        for (int f = 1; f < 64; ++f) {
            float cv = ct[idx];
            float sv = ct[(idx + 96) & 127];
            acc += 2.0f * (Yr[f] * cv - Yi[f] * sv);   // Re(Y e^{+i theta})
            idx = (idx + d) & 127;
        }
        acc *= inv;
        int grow = row0 + row;
        if (grow < N_ITEMS) {
            (m ? outT : outI)[(size_t)grow * DIM + d] = acc;
        }
    }
}

// ---------------------------------------------------------------------------
extern "C" void kernel_launch(void* const* d_in, const int* in_sizes, int n_in,
                              void* d_out, int out_size, void* d_ws, size_t ws_size,
                              hipStream_t stream) {
    // setup_inputs order:
    // 0 image_embeds, 1 text_embeds, 2 image_table, 3 text_table,
    // 4 W_img, 5 b_img, 6 W_txt, 7 b_txt, 8 avrf_img, 9 avrf_txt,
    // 10 avg_R, 11 psi, 12 lambda_weights
    const float* img_table = (const float*)d_in[2];
    const float* txt_table = (const float*)d_in[3];
    const float* W_img     = (const float*)d_in[4];
    const float* b_img     = (const float*)d_in[5];
    const float* W_txt     = (const float*)d_in[6];
    const float* b_txt     = (const float*)d_in[7];
    const float* avrf_img  = (const float*)d_in[8];
    const float* avrf_txt  = (const float*)d_in[9];
    const float* avg_R     = (const float*)d_in[10];
    const float* psi       = (const float*)d_in[11];
    const float* lw        = (const float*)d_in[12];

    bf16_t* WTi = (bf16_t*)d_ws;
    bf16_t* WTt = (bf16_t*)((char*)d_ws + (size_t)RAW_IMG * DIM * sizeof(bf16_t));

    const int totalW = RAW_IMG * DIM + RAW_TXT * DIM;
    prep_wt<<<(totalW + 255) / 256, 256, 0, stream>>>(W_img, W_txt, WTi, WTt);

    const int grid = (N_ITEMS + TILE_M - 1) / TILE_M;
    damps_fused<<<grid, 256, SMEM_BYTES, stream>>>(
        img_table, txt_table, WTi, WTt, b_img, b_txt,
        avrf_img, avrf_txt, avg_R, psi, lw, (float*)d_out);
}
